// PolyAttentionBlock_8693013807186
// MI455X (gfx1250) — matrix operand order repaired and checked by reading, not compiled, not run on hardware
//
#include <hip/hip_runtime.h>

// CDNA5 / gfx1250 (MI455X), wave32. bf16 WMMA 16x16x32 + Tensor Data Mover staging.

typedef __attribute__((ext_vector_type(16))) __bf16 v16bf;
typedef __attribute__((ext_vector_type(8)))  float  v8f;
typedef __attribute__((ext_vector_type(4)))  float  v4f;
typedef unsigned int uint32x4 __attribute__((ext_vector_type(4)));
typedef int          int32x8  __attribute__((ext_vector_type(8)));
typedef int          int32x4  __attribute__((ext_vector_type(4)));

#if __has_builtin(__builtin_amdgcn_tensor_load_to_lds) && __has_builtin(__builtin_amdgcn_s_wait_tensorcnt)
#define USE_TDM 1
#else
#define USE_TDM 0
#endif

// ---------- fp32 -> bf16 (round to nearest even) ----------
__device__ __forceinline__ unsigned short f2bf_bits(float f) {
  union { float f; unsigned u; } in; in.f = f;
  unsigned r = in.u + 0x7FFFu + ((in.u >> 16) & 1u);
  return (unsigned short)(r >> 16);
}
__device__ __forceinline__ __bf16 f2bf(float f) {
  union { unsigned short s; __bf16 b; } o; o.s = f2bf_bits(f);
  return o.b;
}
__device__ __forceinline__ void cvt_store(__bf16* p, float v) { *p = f2bf(v); }
__device__ __forceinline__ void cvt_store(float*  p, float v) { *p = v; }

// ---------- WMMA fragment loaders from LDS (wave32 layouts, ISA 7.12.2) ----------
// A (16x32 bf16): lane L -> row M=L%16; half h=L>>4 -> K groups {h*8.., 16+h*8..}
__device__ __forceinline__ v16bf lds_frag_a(const __bf16* base, int ldm) {
  const int lane = threadIdx.x & 31;
  const __bf16* p = base + (lane & 15) * ldm + (lane >> 4) * 8;
  union { v16bf v; v4f q[2]; } u;
  u.q[0] = *(const v4f*)(p);
  u.q[1] = *(const v4f*)(p + 16);
  return u.v;
}
// B (32x16 bf16) from LDS tile stored [k][n]: lane L -> col N=L%16; half h=L>>4 -> K groups {h*8.., 16+h*8..}
__device__ __forceinline__ v16bf lds_frag_b(const __bf16* base, int ldn) {
  const int lane = threadIdx.x & 31;
  const __bf16* p = base + (lane & 15) + (lane >> 4) * 8 * ldn;
  v16bf v;
#pragma unroll
  for (int s = 0; s < 8; ++s) {
    v[s] = p[s * ldn];
    v[s + 8] = p[(16 + s) * ldn];
  }
  return v;
}

// ---------- Tensor Data Mover: 2D tile (bf16) global -> LDS ----------
#if USE_TDM
__device__ __forceinline__ unsigned lds_addr_of(const void* p) {
  return (unsigned)(size_t)p;   // low 32 bits of generic addr == LDS byte offset
}
// tile_d0 elements contiguous (dim0), tile_d1 rows, row stride `stride` elements.
__device__ __forceinline__ void tdm_load_2d(unsigned lds_byte_addr, const void* gaddr,
                                            unsigned tile_d0, unsigned tile_d1,
                                            unsigned tensor_d0, unsigned stride_d0) {
  unsigned long long ga = (unsigned long long)(size_t)gaddr;
  const unsigned tensor_d1 = 1u << 20;
  uint32x4 g0;
  g0[0] = 1u;                                        // count=1, user desc
  g0[1] = lds_byte_addr;                             // lds_addr
  g0[2] = (unsigned)(ga & 0xFFFFFFFFu);              // global_addr[31:0]
  g0[3] = (unsigned)((ga >> 32) & 0x01FFFFFFu) | (2u << 30);  // addr[56:32], type=2
  int32x8 g1;
  g1[0] = (int)(1u << 16);                                      // data_size=2B
  g1[1] = (int)((tensor_d0 & 0xFFFFu) << 16);                   // tensor_dim0 lo
  g1[2] = (int)((tensor_d0 >> 16) | ((tensor_d1 & 0xFFFFu) << 16)); // dim0 hi | dim1 lo
  g1[3] = (int)((tensor_d1 >> 16) | (tile_d0 << 16));           // dim1 hi | tile_dim0
  g1[4] = (int)(tile_d1 & 0xFFFFu);                             // tile_dim1 (tile_dim2=0)
  g1[5] = (int)stride_d0;                                       // tensor_dim0_stride
  g1[6] = 0;
  g1[7] = 0;
  int32x4 gz4 = {0, 0, 0, 0};
  int32x8 gz8 = {0, 0, 0, 0, 0, 0, 0, 0};
  // 6-arg variant (amdgpu-toolchain clang-23): (g0, g1, g2, g3, g4, cpol)
  __builtin_amdgcn_tensor_load_to_lds(g0, g1, gz4, gz4, gz8, 0);
}
#endif

// ---------- fp32 -> bf16 bulk convert ----------
__global__ void __launch_bounds__(256) cvt_f32_bf16(const float* __restrict__ in,
                                                    __bf16* __restrict__ out, int n4) {
  int i = blockIdx.x * 256 + threadIdx.x;
  if (i >= n4) return;
  v4f f = *(const v4f*)(in + (size_t)i * 4);
  union { __bf16 e[4]; unsigned long long u; } o;
  o.e[0] = f2bf(f.x); o.e[1] = f2bf(f.y); o.e[2] = f2bf(f.z); o.e[3] = f2bf(f.w);
  *(unsigned long long*)(out + (size_t)i * 4) = o.u;
}

// ---------- fp32 [N,K] -> bf16 transposed [K,N] (one-time, weights only) ----------
__global__ void __launch_bounds__(256) cvt_transpose_bf16(const float* __restrict__ in,
                                                          __bf16* __restrict__ out,
                                                          int N, int K) {
  int i = blockIdx.x * 256 + threadIdx.x;
  int total = (N * K) >> 2;
  if (i >= total) return;
  int kq = i % (K >> 2), n = i / (K >> 2);
  v4f f = *(const v4f*)(in + (size_t)n * K + kq * 4);
#pragma unroll
  for (int j = 0; j < 4; ++j) out[(size_t)(kq * 4 + j) * N + n] = f2bf(f[j]);
}

// ---------- GEMM: C[M,N] = A[M,K] @ BT[K,N] + bias[N] ----------
// Block 128(M) x 64(N), BK=32, 8 waves each 32x32 (2x2 WMMA accs).
// TDM double-buffered staging: wave0 DMAs next k-tile while all waves do WMMAs.
template <typename TOut>
__global__ void __launch_bounds__(256) gemm_bf16(
    const __bf16* __restrict__ A, const __bf16* __restrict__ BT,
    const float* __restrict__ bias, TOut* __restrict__ C,
    int M, int N, int K) {
  __shared__ __align__(16) __bf16 As[2][128 * 32];   // [m][k]
  __shared__ __align__(16) __bf16 Bs[2][32 * 64];    // [k][n]

  const int tid = threadIdx.x;
  const int lane = tid & 31;
  const int wid = tid >> 5;
  const int waveM = wid >> 1, waveN = wid & 1;
  const int m0 = blockIdx.y * 128;
  const int n0 = blockIdx.x * 64;

  v8f acc[2][2] = {};
  int cur = 0;

#if USE_TDM
  if (wid == 0) {
    tdm_load_2d(lds_addr_of(&As[0][0]), A + (size_t)m0 * K, 32, 128, K, K);
    tdm_load_2d(lds_addr_of(&Bs[0][0]), BT + n0, 64, 32, N, N);
  }
#endif

  for (int k0 = 0; k0 < K; k0 += 32) {
#if USE_TDM
    if (wid == 0) __builtin_amdgcn_s_wait_tensorcnt(0);
    __syncthreads();                         // tile[cur] visible to all waves
    if (wid == 0 && k0 + 32 < K) {           // prefetch next tile via TDM
      tdm_load_2d(lds_addr_of(&As[cur ^ 1][0]),
                  A + (size_t)m0 * K + (k0 + 32), 32, 128, K, K);
      tdm_load_2d(lds_addr_of(&Bs[cur ^ 1][0]),
                  BT + (size_t)(k0 + 32) * N + n0, 64, 32, N, N);
    }
#else
    __syncthreads();
    { // manual staging (fallback)
      const int ar = tid >> 1, ac = (tid & 1) * 16;
      const __bf16* gA = A + (size_t)(m0 + ar) * K + (k0 + ac);
      *(v4f*)(&As[0][ar * 32 + ac])     = *(const v4f*)(gA);
      *(v4f*)(&As[0][ar * 32 + ac + 8]) = *(const v4f*)(gA + 8);
      const int bk2 = tid >> 3, bn8 = (tid & 7) * 8;
      *(v4f*)(&Bs[0][bk2 * 64 + bn8]) =
          *(const v4f*)(BT + (size_t)(k0 + bk2) * N + n0 + bn8);
    }
    __syncthreads();
#endif

    v16bf af[2], bfr[2];
#pragma unroll
    for (int mi = 0; mi < 2; ++mi)
      af[mi] = lds_frag_a(&As[cur][(waveM * 32 + mi * 16) * 32], 32);
#pragma unroll
    for (int ni = 0; ni < 2; ++ni)
      bfr[ni] = lds_frag_b(&Bs[cur][waveN * 32 + ni * 16], 64);
#pragma unroll
    for (int mi = 0; mi < 2; ++mi)
#pragma unroll
      for (int ni = 0; ni < 2; ++ni)
        acc[mi][ni] = __builtin_amdgcn_wmma_f32_16x16x32_bf16(
            false, af[mi], false, bfr[ni], (short)0, acc[mi][ni], false, false);
#if USE_TDM
    cur ^= 1;
#endif
  }

  const int lh = lane >> 4, ln = lane & 15;
#pragma unroll
  for (int ni = 0; ni < 2; ++ni) {
    const int nn = n0 + waveN * 32 + ni * 16 + ln;
    const float bv = bias[nn];
#pragma unroll
    for (int mi = 0; mi < 2; ++mi)
#pragma unroll
      for (int v = 0; v < 8; ++v) {
        const int mm = m0 + waveM * 32 + mi * 16 + v + 8 * lh;
        cvt_store(&C[(size_t)mm * N + nn], acc[mi][ni][v] + bv);
      }
  }
}

// ---------- Fused flash-style polynomial attention ----------
// Computes S^T = K @ Q^T so K and V both stage in natural [row][d] order (TDM-able,
// no per-tile transpose); only Q is transposed, once per block.
__global__ void __launch_bounds__(256) poly_attn_kernel(
    const __bf16* __restrict__ qkv, const float* __restrict__ poly,
    __bf16* __restrict__ outb) {
  __shared__ __align__(16) __bf16 Qt[64 * 64];     // [d][m]   (Q transposed, once)
  __shared__ __align__(16) __bf16 Ps[64 * 64];     // [m][j]   (poly(S) bf16)
  __shared__ __align__(16) __bf16 Kr[2][64 * 64];  // [j][d]   double-buffered
  __shared__ __align__(16) __bf16 Vs[2][64 * 64];  // [j][d]   double-buffered

  const int tid = threadIdx.x;
  const int lane = tid & 31;
  const int wid = tid >> 5;
  const int waveM = wid >> 1;   // 0..3
  const int waveN = wid & 1;    // 0..1
  const int m0 = blockIdx.x * 64;
  const int bh = blockIdx.y;
  const int b = bh / 12, h = bh % 12;

  const size_t rowbase = (size_t)b * 1024 * 2304;
  const int qoff = h * 64, koff = 768 + h * 64, voff = 1536 + h * 64;
  const float pa = poly[0], pb = poly[1], pc = poly[2];
  const float scale = 0.125f;   // 64^-0.5

  const int sr = tid >> 2;        // 0..63
  const int sc = (tid & 3) * 16;  // 0,16,32,48

  { // stage Q transposed (once per block): Qt[d][m] = Q[m][d]
    union { v4f q[2]; __bf16 e[16]; } uq;
    const __bf16* g = qkv + rowbase + (size_t)(m0 + sr) * 2304 + qoff + sc;
    uq.q[0] = *(const v4f*)(g);
    uq.q[1] = *(const v4f*)(g + 8);
#pragma unroll
    for (int t = 0; t < 16; ++t) Qt[(sc + t) * 64 + sr] = uq.e[t];
  }

  int cur = 0;
#if USE_TDM
  if (wid == 0) {
    tdm_load_2d(lds_addr_of(&Kr[0][0]), qkv + rowbase + koff, 64, 64, 2304, 2304);
    tdm_load_2d(lds_addr_of(&Vs[0][0]), qkv + rowbase + voff, 64, 64, 2304, 2304);
  }
#endif

  v8f oacc[2] = {};

  for (int t = 0; t < 16; ++t) {
#if USE_TDM
    if (wid == 0) __builtin_amdgcn_s_wait_tensorcnt(0);
    __syncthreads();   // K/V[cur] ready; prev iter done with Ps and old buffers
    if (wid == 0 && t + 1 < 16) {
      const size_t roff = rowbase + (size_t)(t + 1) * 64 * 2304;
      tdm_load_2d(lds_addr_of(&Kr[cur ^ 1][0]), qkv + roff + koff, 64, 64, 2304, 2304);
      tdm_load_2d(lds_addr_of(&Vs[cur ^ 1][0]), qkv + roff + voff, 64, 64, 2304, 2304);
    }
#else
    __syncthreads();
    { // manual staging (fallback): both K and V are natural [j][d] copies
      const int j0 = t * 64;
      const __bf16* gk = qkv + rowbase + (size_t)(j0 + sr) * 2304 + koff + sc;
      *(v4f*)(&Kr[0][sr * 64 + sc])     = *(const v4f*)(gk);
      *(v4f*)(&Kr[0][sr * 64 + sc + 8]) = *(const v4f*)(gk + 8);
      const __bf16* gv = qkv + rowbase + (size_t)(j0 + sr) * 2304 + voff + sc;
      *(v4f*)(&Vs[0][sr * 64 + sc])     = *(const v4f*)(gv);
      *(v4f*)(&Vs[0][sr * 64 + sc + 8]) = *(const v4f*)(gv + 8);
    }
    __syncthreads();
#endif

    // S^T = K Q^T : this wave covers 16 j-rows x 32 m-cols, k-dim = d (2 steps)
    v8f sacc[2] = {};
#pragma unroll
    for (int d0 = 0; d0 < 64; d0 += 32) {
      v16bf ak = lds_frag_a(&Kr[cur][(waveM * 16) * 64 + d0], 64);
#pragma unroll
      for (int ni = 0; ni < 2; ++ni) {
        v16bf bq = lds_frag_b(&Qt[d0 * 64 + waveN * 32 + ni * 16], 64);
        sacc[ni] = __builtin_amdgcn_wmma_f32_16x16x32_bf16(
            false, ak, false, bq, (short)0, sacc[ni], false, false);
      }
    }

    // polynomial in fp32; S^T(j,m) -> Ps[m][j]: 8 j-consecutive values per lane
    // pack into one b128 LDS store.
    {
      const int lh = lane >> 4, ln = lane & 15;
#pragma unroll
      for (int ni = 0; ni < 2; ++ni) {
        const int m = waveN * 32 + ni * 16 + ln;
        union { __bf16 e[8]; v4f q; } up;
#pragma unroll
        for (int v = 0; v < 8; ++v) {
          float s = sacc[ni][v] * scale;
          up.e[v] = f2bf((pa * s + pb) * s + pc);
        }
        *(v4f*)(&Ps[m * 64 + waveM * 16 + 8 * lh]) = up.q;
      }
    }
    __syncthreads();

    // O += P V : this wave covers 16 m-rows x 32 d-cols, k-dim = j (2 steps)
#pragma unroll
    for (int jk = 0; jk < 64; jk += 32) {
      v16bf ap = lds_frag_a(&Ps[(waveM * 16) * 64 + jk], 64);
#pragma unroll
      for (int di = 0; di < 2; ++di) {
        v16bf bv = lds_frag_b(&Vs[cur][jk * 64 + waveN * 32 + di * 16], 64);
        oacc[di] = __builtin_amdgcn_wmma_f32_16x16x32_bf16(
            false, ap, false, bv, (short)0, oacc[di], false, false);
      }
    }
#if USE_TDM
    cur ^= 1;
#endif
  }

  // write O tile: [B,N,H,D] merged into [B*N, 768]
  const int lh = lane >> 4, ln = lane & 15;
#pragma unroll
  for (int di = 0; di < 2; ++di)
#pragma unroll
    for (int v = 0; v < 8; ++v) {
      const int mm = m0 + waveM * 16 + v + 8 * lh;
      const int dd = waveN * 32 + di * 16 + ln;
      outb[((size_t)b * 1024 + mm) * 768 + h * 64 + dd] = f2bf(oacc[di][v]);
    }
}

// ---------------------------------------------------------------------------
extern "C" void kernel_launch(void* const* d_in, const int* in_sizes, int n_in,
                              void* d_out, int out_size, void* d_ws, size_t ws_size,
                              hipStream_t stream) {
  (void)in_sizes; (void)n_in; (void)out_size; (void)ws_size;
  const float* x      = (const float*)d_in[0];   // [16,1024,768]
  const float* w_qkv  = (const float*)d_in[1];   // [2304,768]
  const float* b_qkv  = (const float*)d_in[2];   // [2304]
  const float* w_proj = (const float*)d_in[3];   // [768,768]
  const float* b_proj = (const float*)d_in[4];   // [768]
  const float* poly   = (const float*)d_in[5];   // [3]
  float* out = (float*)d_out;                    // [16,1024,768] fp32

  const int M = 16 * 1024;
  const int C = 768, C3 = 2304;

  // workspace (bf16): x | w_qkv^T | w_proj^T | qkv | attn_out
  __bf16* xb     = (__bf16*)d_ws;                  // M*C
  __bf16* wqkvT  = xb    + (size_t)M * C;          // [C][C3]
  __bf16* wprojT = wqkvT + (size_t)C * C3;         // [C][C]
  __bf16* qkvb   = wprojT + (size_t)C * C;         // M*C3
  __bf16* attnb  = qkvb  + (size_t)M * C3;         // M*C

  // 1) conversions (weights transposed once -> no per-tile transposes in GEMMs)
  cvt_f32_bf16<<<(M * C / 4 + 255) / 256, 256, 0, stream>>>(x, xb, M * C / 4);
  cvt_transpose_bf16<<<(C3 * C / 4 + 255) / 256, 256, 0, stream>>>(w_qkv, wqkvT, C3, C);
  cvt_transpose_bf16<<<(C * C / 4 + 255) / 256, 256, 0, stream>>>(w_proj, wprojT, C, C);

  // 2) qkv = x @ w_qkv^T + b_qkv (bf16 out)
  gemm_bf16<__bf16><<<dim3(C3 / 64, M / 128), 256, 0, stream>>>(
      xb, wqkvT, b_qkv, qkvb, M, C3, C);

  // 3) fused polynomial attention (flash-style, TDM double-buffered K/V)
  poly_attn_kernel<<<dim3(1024 / 64, 16 * 12), 256, 0, stream>>>(qkvb, poly, attnb);

  // 4) out = attn @ w_proj^T + b_proj (fp32 out)
  gemm_bf16<float><<<dim3(C / 64, M / 128), 256, 0, stream>>>(
      attnb, wprojT, b_proj, out, M, C, C);
}
